// MaterialGraphLayer_47974784696416
// MI455X (gfx1250) — matrix operand order repaired and checked by reading, not compiled, not run on hardware
//
#include <hip/hip_runtime.h>

#define N_NODES 25000
#define N_EDGES 300000
#define D_NODE  256
#define D_EDGE  64
#define D_IN    320            // D_NODE + D_EDGE
#define KTILES  10             // 320 / 32
#define NTILES  16             // 256 / 16
#define ETILES  (N_EDGES / 16) // 18750 exact
#define LN_EPS  1e-3f

typedef __attribute__((ext_vector_type(16))) __bf16 v16bf;
typedef __attribute__((ext_vector_type(8)))  float  v8f;
typedef __attribute__((ext_vector_type(4)))  float  v4f;

// ---------------------------------------------------------------------------
// Kernel 1: repack W (320x256 f32, row-major) into bf16 in WMMA B-fragment
// order. For tile (kt, nt): lane = ((k%32)/16)*16 + (n%16), element = k%16.
// Each (kt,nt) tile = 512 bf16 contiguous -> lane reads 16 contiguous bf16.
// Needs 320*256*2 = 163840 bytes of d_ws.
// ---------------------------------------------------------------------------
__global__ void repack_W(const float* __restrict__ W,
                         unsigned short* __restrict__ Wp) {
    int idx = blockIdx.x * blockDim.x + threadIdx.x;
    if (idx >= D_IN * D_NODE) return;
    int k = idx / D_NODE;
    int n = idx % D_NODE;
    int kt   = k >> 5;
    int kk   = k & 31;
    int half = kk >> 4;
    int e    = kk & 15;
    int nt   = n >> 4;
    int nn   = n & 15;
    int lane = half * 16 + nn;
    int dst  = ((kt * NTILES + nt) << 9) + lane * 16 + e;
    __bf16 v = (__bf16)W[idx];
    Wp[dst] = __builtin_bit_cast(unsigned short, v);
}

// ---------------------------------------------------------------------------
// Kernel 2: zero the f32 accumulator (d_out) with 16B stores.
// N_NODES*D_NODE = 6.4M floats = 1.6M v4f; 6250 blocks * 256 threads exact.
// ---------------------------------------------------------------------------
__global__ void zero_out(float* __restrict__ out) {
    long i = (long)(blockIdx.x * blockDim.x + threadIdx.x);
    v4f z = (v4f)0.0f;
    ((v4f*)out)[i] = z;
}

// ---------------------------------------------------------------------------
// Kernel 3: edge GEMM (bf16 WMMA, f32 acc) + bias + SiLU + atomic scatter-add.
// Block = 128 threads = 4 wave32; wave w handles edge tile (blockIdx*4 + w).
// Per wave: preload 10 A fragments (16 edges x 32 K in bf16), then loop over
// 16 N-tiles: 10x v_wmma_f32_16x16x32_bf16, epilogue, 8 f32 atomics/lane.
// ---------------------------------------------------------------------------
__global__ __launch_bounds__(128)
void edge_gemm_scatter(const float* __restrict__ nodef,
                       const int*   __restrict__ eidx,
                       const float* __restrict__ edgef,
                       const unsigned short* __restrict__ Wp,
                       const float* __restrict__ bias,
                       float* __restrict__ out) {
    const int wave = threadIdx.x >> 5;
    const int lane = threadIdx.x & 31;
    const int tt   = blockIdx.x * 4 + wave;   // 16-edge tile id
    if (tt >= ETILES) return;                  // uniform per-wave exit

    const int mlo  = lane & 15;                // A-row handled by this lane
    const int half = lane >> 4;                // K-half select (contiguous k order)

    const int  e_a    = tt * 16 + mlo;
    const int  srcRow = eidx[e_a];             // edge_index[0][e]
    const float* __restrict__ arow = nodef + (long)srcRow * D_NODE;
    const float* __restrict__ erow = edgef + (long)e_a * D_EDGE;

    // ---- preload all 10 A fragments, converting f32 -> bf16 -------------
    // elements 0..15 : K = kb + half*16 + i
    v16bf afr[KTILES];
#pragma unroll
    for (int kt = 0; kt < KTILES; ++kt) {
        const int kb = kt * 32;
        const float* srcp = (kb < D_NODE) ? (arow + kb) : (erow + (kb - D_NODE));
        v4f x0 = *(const v4f*)(srcp + half * 16);
        v4f x1 = *(const v4f*)(srcp + half * 16 + 4);
        v4f x2 = *(const v4f*)(srcp + half * 16 + 8);
        v4f x3 = *(const v4f*)(srcp + half * 16 + 12);
        v16bf a;
        a[0]  = (__bf16)x0[0]; a[1]  = (__bf16)x0[1];
        a[2]  = (__bf16)x0[2]; a[3]  = (__bf16)x0[3];
        a[4]  = (__bf16)x1[0]; a[5]  = (__bf16)x1[1];
        a[6]  = (__bf16)x1[2]; a[7]  = (__bf16)x1[3];
        a[8]  = (__bf16)x2[0]; a[9]  = (__bf16)x2[1];
        a[10] = (__bf16)x2[2]; a[11] = (__bf16)x2[3];
        a[12] = (__bf16)x3[0]; a[13] = (__bf16)x3[1];
        a[14] = (__bf16)x3[2]; a[15] = (__bf16)x3[3];
        afr[kt] = a;
    }

    // ---- scatter targets for the 8 output rows this lane owns -----------
    // C/D layout: VGPR r, lane l -> M = (l/16)*8 + r, N = (l%16)
    int tgt[8];
#pragma unroll
    for (int r = 0; r < 8; ++r)
        tgt[r] = eidx[N_EDGES + tt * 16 + half * 8 + r];

    // ---- N-tile loop: WMMA chain + epilogue ------------------------------
    const unsigned short* __restrict__ wl = Wp + lane * 16;
    for (int nt = 0; nt < NTILES; ++nt) {
        v8f c = (v8f)0.0f;
#pragma unroll
        for (int kt = 0; kt < KTILES; ++kt) {
            v16bf bfr = *(const v16bf*)(wl + ((kt * NTILES + nt) << 9));
            c = __builtin_amdgcn_wmma_f32_16x16x32_bf16(
                    false, afr[kt], false, bfr, (short)0, c, false, false);
        }
        const int   n  = nt * 16 + mlo;
        const float bn = bias[n];
#pragma unroll
        for (int r = 0; r < 8; ++r) {
            float v = c[r] + bn;
            float s = v * (1.0f / (1.0f + __expf(-v)));   // SiLU
            __hip_atomic_fetch_add(out + (long)tgt[r] * D_NODE + n, s,
                                   __ATOMIC_RELAXED, __HIP_MEMORY_SCOPE_AGENT);
        }
    }
}

// ---------------------------------------------------------------------------
// Kernel 4: row-wise LayerNorm, in place on d_out. One block per node row,
// 256 threads (one per column), LDS tree reduction for mean and variance.
// ---------------------------------------------------------------------------
__global__ __launch_bounds__(256)
void layernorm_rows(float* __restrict__ out,
                    const float* __restrict__ gamma,
                    const float* __restrict__ beta) {
    __shared__ float red[256];
    const int  row = blockIdx.x;
    const int  n   = threadIdx.x;
    float x = out[(long)row * D_NODE + n];

    red[n] = x;
    __syncthreads();
#pragma unroll
    for (int s = 128; s > 0; s >>= 1) {
        if (n < s) red[n] += red[n + s];
        __syncthreads();
    }
    const float mean = red[0] * (1.0f / D_NODE);
    __syncthreads();

    const float d = x - mean;
    red[n] = d * d;
    __syncthreads();
#pragma unroll
    for (int s = 128; s > 0; s >>= 1) {
        if (n < s) red[n] += red[n + s];
        __syncthreads();
    }
    const float var = red[0] * (1.0f / D_NODE);

    out[(long)row * D_NODE + n] = d * rsqrtf(var + LN_EPS) * gamma[n] + beta[n];
}

// ---------------------------------------------------------------------------
// Launch. Input order: node_features, edge_index, edge_features, W, b,
// gamma, beta. Output: 25000x256 f32. d_ws: 160KB for repacked bf16 W.
// ---------------------------------------------------------------------------
extern "C" void kernel_launch(void* const* d_in, const int* in_sizes, int n_in,
                              void* d_out, int out_size, void* d_ws, size_t ws_size,
                              hipStream_t stream) {
    const float* nodef = (const float*)d_in[0];
    const int*   eidx  = (const int*)  d_in[1];
    const float* edgef = (const float*)d_in[2];
    const float* W     = (const float*)d_in[3];
    const float* bias  = (const float*)d_in[4];
    const float* gamma = (const float*)d_in[5];
    const float* beta  = (const float*)d_in[6];
    float* out = (float*)d_out;
    unsigned short* Wp = (unsigned short*)d_ws;

    // 1) repack W into WMMA B-fragment bf16 layout (320*256 threads)
    hipLaunchKernelGGL(repack_W, dim3((D_IN * D_NODE + 255) / 256), dim3(256),
                       0, stream, W, Wp);
    // 2) zero the accumulator (6.4M floats / 4 per thread)
    hipLaunchKernelGGL(zero_out, dim3(N_NODES * D_NODE / 4 / 256), dim3(256),
                       0, stream, out);
    // 3) edge GEMM + SiLU + scatter-add (4 wave-tiles of 16 edges per block)
    hipLaunchKernelGGL(edge_gemm_scatter, dim3((ETILES + 3) / 4), dim3(128),
                       0, stream, nodef, eidx, edgef, Wp, bias, out);
    // 4) LayerNorm per node row
    hipLaunchKernelGGL(layernorm_rows, dim3(N_NODES), dim3(256),
                       0, stream, out, gamma, beta);
}
